// BaseImplicitConv_3118146257429
// MI455X (gfx1250) — matrix operand order repaired and checked by reading, not compiled, not run on hardware
//
#include <hip/hip_runtime.h>
#include <hip/hip_bf16.h>

#define B_   4
#define L_   4096
#define D_   1024
#define DH_  16
#define NFFT 8192
#define PI_F 3.14159265358979323846f

typedef __attribute__((ext_vector_type(16))) __bf16 v16bf;
typedef __attribute__((ext_vector_type(8)))  float  v8f;

// ---------------------------------------------------------------- filter MLP
__global__ void k_mlp1(const float* __restrict__ z, const float* __restrict__ w1,
                       const float* __restrict__ b1, float* __restrict__ h) {
    int l = blockIdx.x * blockDim.x + threadIdx.x;
    if (l >= L_) return;
    float p0 = z[l * 3 + 0], p1 = z[l * 3 + 1], p2 = z[l * 3 + 2];
#pragma unroll
    for (int j = 0; j < DH_; ++j) {
        float v = fmaf(p0, w1[j * 3 + 0], fmaf(p1, w1[j * 3 + 1], fmaf(p2, w1[j * 3 + 2], b1[j])));
        h[l * DH_ + j] = fmaxf(v, 0.0f);
    }
}

// filtT[d][l] = sum_h h[l][hh]*w2[d][hh] + b2[d]
__global__ void k_mlp2(const float* __restrict__ h, const float* __restrict__ w2,
                       const float* __restrict__ b2, float* __restrict__ filtT) {
    int idx = blockIdx.x * blockDim.x + threadIdx.x;   // idx = d*L + l
    int d = idx >> 12, l = idx & (L_ - 1);
    float acc = b2[d];
    const float* hr = h + l * DH_;
    const float* wr = w2 + d * DH_;
#pragma unroll
    for (int j = 0; j < DH_; ++j) acc = fmaf(hr[j], wr[j], acc);
    filtT[idx] = acc;
}

// ---------------------------------------------------------------- sync helpers
__device__ inline void wait_async0() {
#if __has_builtin(__builtin_amdgcn_s_wait_asynccnt)
    __builtin_amdgcn_s_wait_asynccnt(0);
#else
    asm volatile("s_wait_asynccnt 0" ::: "memory");
#endif
}

// ---------------------------------------------------------------- FFT core (split re/im planes + LDS twiddle table)
// tw[k] = exp(-2*pi*i*k/NFFT), k in [0, NFFT/2)

// In-place DIF radix-2: natural in -> bit-reversed out (forward)
__device__ void fft_dif(float* re, float* im, const float2* tw, int tid, int nth) {
    int sh = 0;
    for (int len = NFFT; len >= 2; len >>= 1, ++sh) {
        int half = len >> 1;
        for (int i = tid; i < NFFT / 2; i += nth) {
            int j = i & (half - 1);
            int i0 = ((i & ~(half - 1)) << 1) + j;
            int i1 = i0 + half;
            float ar = re[i0], ai = im[i0];
            float br = re[i1], bi = im[i1];
            float2 w = tw[j << sh];
            float dr = ar - br, di = ai - bi;
            re[i0] = ar + br;             im[i0] = ai + bi;
            re[i1] = dr * w.x - di * w.y; im[i1] = dr * w.y + di * w.x;
        }
        __syncthreads();
    }
}

// In-place DIT radix-2 with conjugate twiddles: bit-reversed in -> natural out (unscaled inverse)
__device__ void fft_dit_inv(float* re, float* im, const float2* tw, int tid, int nth) {
    int sh = 12;
    for (int len = 2; len <= NFFT; len <<= 1, --sh) {
        int half = len >> 1;
        for (int i = tid; i < NFFT / 2; i += nth) {
            int j = i & (half - 1);
            int i0 = ((i & ~(half - 1)) << 1) + j;
            int i1 = i0 + half;
            float2 w = tw[j << sh];                  // conj applied below
            float br = re[i1], bi = im[i1];
            float cr = br * w.x + bi * w.y;          // b * conj(w)
            float ci = bi * w.x - br * w.y;
            float ar = re[i0], ai = im[i0];
            re[i0] = ar + cr; im[i0] = ai + ci;
            re[i1] = ar - cr; im[i1] = ai - ci;
        }
        __syncthreads();
    }
}

// Async-DMA the 4096-float row into re[0..L-1]; overlap twiddle build + zero fill.
__device__ inline void fft_prologue(const float* __restrict__ row,
                                    float* re, float* im, float2* tw, int tid, int nth) {
    for (int o = tid * 16; o < L_ * 4; o += nth * 16) {          // 16B per lane per iter
        unsigned    lo = (unsigned)(uintptr_t)((const char*)re + o);
        const void* ga = (const char*)row + o;
        asm volatile("global_load_async_to_lds_b128 %0, %1, off" :: "v"(lo), "v"(ga) : "memory");
    }
    const float w0 = -2.0f * PI_F / (float)NFFT;                  // build table while DMA flies
    for (int k = tid; k < NFFT / 2; k += nth) {
        float s, c;
        __sincosf(w0 * (float)k, &s, &c);
        tw[k] = make_float2(c, s);
    }
    for (int i = tid; i < NFFT; i += nth) im[i] = 0.0f;
    for (int i = L_ + tid; i < NFFT; i += nth) re[i] = 0.0f;
    wait_async0();
    __syncthreads();
}

// Per-channel filter spectrum (bit-reversed domain), 1/N folded in.
__global__ void k_filt_fft(const float* __restrict__ filtT, float* __restrict__ kf) {
    extern __shared__ float smem[];
    float*  re = smem;                       // NFFT floats
    float*  im = smem + NFFT;                // NFFT floats
    float2* tw = (float2*)(smem + 2 * NFFT); // NFFT/2 float2
    int d = blockIdx.x, tid = threadIdx.x, nth = blockDim.x;
    fft_prologue(filtT + (size_t)d * L_, re, im, tw, tid, nth);
    fft_dif(re, im, tw, tid, nth);
    const float scale = 1.0f / (float)NFFT;
    float2* out = (float2*)(kf + (size_t)d * NFFT * 2);
    for (int i = tid; i < NFFT; i += nth)
        out[i] = make_float2(re[i] * scale, im[i] * scale);
}

// ---------------------------------------------------------------- transpose u -> uT (B,D,L)
__global__ void k_transpose(const float* __restrict__ u, float* __restrict__ uT) {
    __shared__ float tile[32][33];
    int b = blockIdx.z;
    int l0 = blockIdx.y * 32, d0 = blockIdx.x * 32;
    const float* ub = u + (size_t)b * L_ * D_;
    for (int r = threadIdx.y; r < 32; r += 8)
        tile[r][threadIdx.x] = ub[(size_t)(l0 + r) * D_ + d0 + threadIdx.x];
    __syncthreads();
    float* ob = uT + (size_t)b * D_ * L_;
    for (int r = threadIdx.y; r < 32; r += 8)
        ob[(size_t)(d0 + r) * L_ + l0 + threadIdx.x] = tile[threadIdx.x][r];
}

// ---------------------------------------------------------------- per-row FFT convolution (in place)
__global__ void k_conv(float* __restrict__ uT, const float* __restrict__ kf) {
    extern __shared__ float smem[];
    float*  re = smem;
    float*  im = smem + NFFT;
    float2* tw = (float2*)(smem + 2 * NFFT);
    int bd = blockIdx.x;                 // b*D + d
    int d = bd & (D_ - 1);
    int tid = threadIdx.x, nth = blockDim.x;
    float* row = uT + (size_t)bd * L_;

    fft_prologue(row, re, im, tw, tid, nth);
    fft_dif(re, im, tw, tid, nth);                    // bit-reversed U
    const float2* kr = (const float2*)(kf + (size_t)d * NFFT * 2);
    for (int i = tid; i < NFFT; i += nth) {           // both spectra bit-reversed: aligned
        float2 k = kr[i];
        float xr = re[i], xi = im[i];
        re[i] = xr * k.x - xi * k.y;
        im[i] = xr * k.y + xi * k.x;
    }
    __syncthreads();
    fft_dit_inv(re, im, tw, tid, nth);                // natural-order y (scale already in kf)

    for (int o = tid * 16; o < L_ * 4; o += nth * 16) {   // DMA result back out of LDS
        unsigned lo = (unsigned)(uintptr_t)((const char*)re + o);
        void*    ga = (char*)row + o;
        asm volatile("global_store_async_from_lds_b128 %0, %1, off" :: "v"(ga), "v"(lo) : "memory");
    }
    wait_async0();
}

// ---------------------------------------------------------------- WMMA GEMM + fused epilogue
// out[m,n] = (sum_k u[m,k]*pw[n,k] + pb[n]) * y[b,n,l] + u[m,n],  m = b*L + l
__global__ void k_gemm_epi(const float* __restrict__ u, const float* __restrict__ pw,
                           const float* __restrict__ pb, const float* __restrict__ yT,
                           float* __restrict__ out) {
    int lane = threadIdx.x & 31;
    int wv   = threadIdx.x >> 5;
    int tn = blockIdx.x * 16;                   // output-channel tile base
    int tm = (blockIdx.y * 4 + wv) * 16;        // row tile base (b*L + l)
    int half = lane >> 4;                       // lane group 0/1
    int lid  = lane & 15;

    const float* Arow = u  + (size_t)(tm + lid) * D_;   // A row, K contiguous
    const float* Brow = pw + (size_t)(tn + lid) * D_;   // B col n=tn+lid, K contiguous
    v8f acc = {};

#pragma unroll 4
    for (int k0 = 0; k0 < D_; k0 += 32) {
        __builtin_prefetch(Arow + k0 + 128, 0, 1);      // global_prefetch_b8
        __builtin_prefetch(Brow + k0 + 128, 0, 1);
        // A 16x32 bf16 lane map: grp0 e0-7 -> K+0..7, e8-15 -> K+16..23; grp1 shifted by 8
        int ka = k0 + half * 8;
        float4 a0 = *(const float4*)(Arow + ka);
        float4 a1 = *(const float4*)(Arow + ka + 4);
        float4 a2 = *(const float4*)(Arow + ka + 16);
        float4 a3 = *(const float4*)(Arow + ka + 20);
        // B 32x16 bf16 lane map: grp0 e0-7 -> K+0..7, e8-15 -> K+16..23; grp1 shifted by 8
        int kb = k0 + half * 8;
        float4 b0 = *(const float4*)(Brow + kb);
        float4 b1 = *(const float4*)(Brow + kb + 4);
        float4 b2 = *(const float4*)(Brow + kb + 16);
        float4 b3 = *(const float4*)(Brow + kb + 20);
        v16bf av, bv;
        av[0]=(__bf16)a0.x; av[1]=(__bf16)a0.y; av[2]=(__bf16)a0.z; av[3]=(__bf16)a0.w;
        av[4]=(__bf16)a1.x; av[5]=(__bf16)a1.y; av[6]=(__bf16)a1.z; av[7]=(__bf16)a1.w;
        av[8]=(__bf16)a2.x; av[9]=(__bf16)a2.y; av[10]=(__bf16)a2.z; av[11]=(__bf16)a2.w;
        av[12]=(__bf16)a3.x; av[13]=(__bf16)a3.y; av[14]=(__bf16)a3.z; av[15]=(__bf16)a3.w;
        bv[0]=(__bf16)b0.x; bv[1]=(__bf16)b0.y; bv[2]=(__bf16)b0.z; bv[3]=(__bf16)b0.w;
        bv[4]=(__bf16)b1.x; bv[5]=(__bf16)b1.y; bv[6]=(__bf16)b1.z; bv[7]=(__bf16)b1.w;
        bv[8]=(__bf16)b2.x; bv[9]=(__bf16)b2.y; bv[10]=(__bf16)b2.z; bv[11]=(__bf16)b2.w;
        bv[12]=(__bf16)b3.x; bv[13]=(__bf16)b3.y; bv[14]=(__bf16)b3.z; bv[15]=(__bf16)b3.w;
        acc = __builtin_amdgcn_wmma_f32_16x16x32_bf16(false, av, false, bv,
                                                      (short)0, acc, false, false);
    }

    // Epilogue. C/D layout: lane grp g, VGPR j -> M = tm + j + 8g, N = tn + lid.
    int bb = tm >> 12;                   // batch (tiles never cross batch: 4096 % 16 == 0)
    int l0 = tm & (L_ - 1);
    const float* yrow = yT + ((size_t)(bb * D_ + tn + lid)) * L_ + l0;
    const float* ucol = u   + (size_t)tm * D_ + (tn + lid);
    float*       ocol = out + (size_t)tm * D_ + (tn + lid);
    float bias = pb[tn + lid];
#pragma unroll
    for (int j = 0; j < 8; ++j) {
        int r = j + half * 8;
        float proj = acc[j] + bias;
        ocol[(size_t)r * D_] = yrow[r] * proj + ucol[(size_t)r * D_];
    }
}

// ---------------------------------------------------------------- launch
extern "C" void kernel_launch(void* const* d_in, const int* in_sizes, int n_in,
                              void* d_out, int out_size, void* d_ws, size_t ws_size,
                              hipStream_t stream) {
    const float* u  = (const float*)d_in[0];
    const float* z  = (const float*)d_in[1];
    const float* w1 = (const float*)d_in[2];
    const float* b1 = (const float*)d_in[3];
    const float* w2 = (const float*)d_in[4];
    const float* b2 = (const float*)d_in[5];
    const float* pw = (const float*)d_in[6];
    const float* pb = (const float*)d_in[7];
    float* out = (float*)d_out;

    float* ws    = (float*)d_ws;
    float* h     = ws;                       // 4096*16            = 65,536 f
    float* filtT = h + (size_t)L_ * DH_;     // 1024*4096          = 4,194,304 f
    float* kf    = filtT + (size_t)D_ * L_;  // 1024*8192*2        = 16,777,216 f
    float* uT    = kf + (size_t)D_ * NFFT*2; // 4*1024*4096 (y in place) = 16,777,216 f

    size_t lds_fft = (size_t)(2 * NFFT) * sizeof(float) + (NFFT / 2) * sizeof(float2); // 96 KB

    k_mlp1<<<L_ / 256, 256, 0, stream>>>(z, w1, b1, h);
    k_mlp2<<<(D_ * L_) / 256, 256, 0, stream>>>(h, w2, b2, filtT);
    k_filt_fft<<<D_, 512, lds_fft, stream>>>(filtT, kf);
    k_transpose<<<dim3(D_ / 32, L_ / 32, B_), dim3(32, 8), 0, stream>>>(u, uT);
    k_conv<<<B_ * D_, 512, lds_fft, stream>>>(uT, kf);
    k_gemm_epi<<<dim3(D_ / 16, (B_ * L_) / (16 * 4)), 128, 0, stream>>>(u, pw, pb, uT, out);
}
